// MVP_CrossAttn_64759516889866
// MI455X (gfx1250) — matrix operand order repaired and checked by reading, not compiled, not run on hardware
//
#include <hip/hip_runtime.h>
#include <hip/hip_bf16.h>
#include <math.h>

#define B_ 4
#define N_ 511
#define S_ 512
#define H_ 128
#define W_ 128
#define D_ 256
#define K_ 4096
#define GS_ 64
#define HEADS_ 8
#define HD_ 32
#define P_ 9
#define R_ 4
#define PP_ 96                     /* patch features 81 padded to 96 */
#define SCALE_ 0.1767766952966369f /* 32^-0.5 */
#define INV2SIG2_ 78.125f          /* 1/(2*0.08^2) */

typedef __attribute__((ext_vector_type(16))) _Float16 v16h;
typedef __attribute__((ext_vector_type(8)))  _Float16 h8;
typedef __attribute__((ext_vector_type(4)))  _Float16 h4;
typedef __attribute__((ext_vector_type(8)))  float    v8f;

union FragH { v16h v; h8 h[2]; _Float16 e[16]; };

__device__ __forceinline__ float gelu_f(float x) {
  return 0.5f * x * (1.0f + erff(x * 0.7071067811865476f));
}

// A fragment 16x32 f16: base -> row 0 of a 16-row strip, 32 halfs/row (row-major).
// ISA layout: lanes 0-15 row M=lane, halfs = K 0..7 / 16..23; lanes 16-31 K +8.
__device__ __forceinline__ v16h ldsA_frag(const _Float16* base) {
  int lane = threadIdx.x & 31;
  const _Float16* p = base + (lane & 15) * 32;
  int kb = (lane >> 4) << 3;
  FragH f;
  f.h[0] = *(const h8*)(p + kb);
  f.h[1] = *(const h8*)(p + 16 + kb);
  return f.v;
}

// B fragment 32x16 f16 stored n-major in LDS ([n][k], 32 halfs/row); base -> row n0.
// ISA layout: lanes 0-15 N=lane, halfs = K 0..7 / 16..23; lanes 16-31 K +8.
__device__ __forceinline__ v16h ldsB_frag(const _Float16* base) {
  int lane = threadIdx.x & 31;
  const _Float16* p = base + (lane & 15) * 32 + ((lane >> 4) << 3);
  FragH f;
  f.h[0] = *(const h8*)(p);
  f.h[1] = *(const h8*)(p + 16);
  return f.v;
}

// one 64x64x32 WMMA macro-step from staged LDS tiles
__device__ __forceinline__ void wmma_step(const _Float16* ldsA, const _Float16* ldsB,
                                          int wave, v8f* acc) {
  v16h af = ldsA_frag(ldsA + wave * 16 * 32);
#pragma unroll
  for (int ns = 0; ns < 4; ++ns) {
    v16h bf = ldsB_frag(ldsB + ns * 16 * 32);
    acc[ns] = __builtin_amdgcn_wmma_f32_16x16x32_f16(false, af, false, bf,
                                                     (short)0, acc[ns], false, false);
  }
}

// ---------------------------------------------------------------------------
// Generic tiled WMMA GEMM: C = epi(A[MxKd] @ W[KwxNn] + bias)
// Requirements: Kd % 32 == 0, Nn % 64 == 0 (holds for all call sites).
// mode 0: standard (optional gelu, optional residual add after act)
// mode 1: measurement-token epilogue (adds meas linear, row remap bn -> b*S+n)
// mode 2: per-batch task weights (W = tfp_W[task_id[b]], bias = tfp_b[task_id[b]])
// ---------------------------------------------------------------------------
__launch_bounds__(128)
__global__ void gemm_kernel(const float* __restrict__ A, const float* __restrict__ W,
                            const float* __restrict__ bias, float* __restrict__ C,
                            const float* __restrict__ resid,
                            int M, int Nn, int Kd, int Kw, int act_gelu, int mode,
                            const float* __restrict__ meas_xy, const float* __restrict__ meas_v,
                            const float* __restrict__ meas_W, const float* __restrict__ meas_b,
                            const int* __restrict__ task_id)
{
  __shared__ __align__(16) _Float16 ldsA[64 * 32];
  __shared__ __align__(16) _Float16 ldsB[64 * 32];  // n-major: [n][k]
  const int tid  = threadIdx.x;
  const int wave = tid >> 5;
  const int lane = tid & 31;
  const int m0 = blockIdx.y * 64;
  const int n0 = blockIdx.x * 64;

  const float* Wp = W;
  const float* bp = bias;
  if (mode == 2) {
    int t = task_id[m0 >> 12];              // rows-per-batch = K_ = 4096
    Wp += (size_t)t * Kd * Nn;
    bp += (size_t)t * Nn;
  }

  // B-tile 4x4 transpose block owned by this thread
  const int bkb = (tid & 7) * 4;            // k offset within tile
  const int bnb = (tid >> 3) * 4;           // n offset within tile
  _Float16* const sbB = ldsB + bnb * 32 + bkb;   // loop-invariant LDS store bases
  _Float16* const sbA = ldsA + tid * 4;

  v8f acc[4] = {};

  const bool fullA = (m0 + 64 <= M);

  if (fullA && Kw == Kd) {
    // ---- fast path: interior tiles, pointer-increment addressing ----
    const float* pa = A + (size_t)(m0 + (tid >> 3)) * Kd + (tid & 7) * 4;
    const float* pb = Wp + (size_t)bkb * Nn + n0 + bnb;
    const size_t aStride = (size_t)16 * Kd;     // rows between the 4 A sub-loads
    for (int k0 = 0; k0 < Kd; k0 += 32) {
#pragma unroll
      for (int it = 0; it < 4; ++it) {
        float4 w = *(const float4*)(pa + aStride * it);
        h4 hv = { (_Float16)w.x, (_Float16)w.y, (_Float16)w.z, (_Float16)w.w };
        *(h4*)(sbA + it * 512) = hv;
      }
      float4 w[4];
#pragma unroll
      for (int j = 0; j < 4; ++j) w[j] = *(const float4*)(pb + (size_t)j * Nn);
#pragma unroll
      for (int nj = 0; nj < 4; ++nj) {
        h4 hv = { (_Float16)((&w[0].x)[nj]), (_Float16)((&w[1].x)[nj]),
                  (_Float16)((&w[2].x)[nj]), (_Float16)((&w[3].x)[nj]) };
        *(h4*)(sbB + nj * 32) = hv;
      }
      __syncthreads();
      if (k0 + 32 < Kd)
        __builtin_prefetch(pb + (size_t)32 * Nn, 0, 1);
      wmma_step(ldsA, ldsB, wave, acc);
      __syncthreads();
      pa += 32;
      pb += (size_t)32 * Nn;
    }
  } else {
    // ---- general path: clamped, branch-free fills (patch GEMM edge tiles) ----
    for (int k0 = 0; k0 < Kd; k0 += 32) {
#pragma unroll
      for (int it = 0; it < 4; ++it) {
        int i = it * 512 + tid * 4;
        int mm = i >> 5, kk = i & 31;
        int gm = m0 + mm;
        int gmc = min(gm, M - 1);
        float4 w = *(const float4*)(A + (size_t)gmc * Kd + k0 + kk);
        if (gm >= M) { w.x = 0.f; w.y = 0.f; w.z = 0.f; w.w = 0.f; }
        h4 hv = { (_Float16)w.x, (_Float16)w.y, (_Float16)w.z, (_Float16)w.w };
        *(h4*)(ldsA + i) = hv;
      }
      {
        float4 w[4];
#pragma unroll
        for (int j = 0; j < 4; ++j) {
          int gk = k0 + bkb + j;
          int gkc = min(gk, Kw - 1);
          w[j] = *(const float4*)(Wp + (size_t)gkc * Nn + n0 + bnb);
          if (gk >= Kw) { w[j].x = 0.f; w[j].y = 0.f; w[j].z = 0.f; w[j].w = 0.f; }
        }
#pragma unroll
        for (int nj = 0; nj < 4; ++nj) {
          h4 hv = { (_Float16)((&w[0].x)[nj]), (_Float16)((&w[1].x)[nj]),
                    (_Float16)((&w[2].x)[nj]), (_Float16)((&w[3].x)[nj]) };
          *(h4*)(sbB + nj * 32) = hv;
        }
      }
      __syncthreads();
      wmma_step(ldsA, ldsB, wave, acc);
      __syncthreads();
    }
  }

  const int cn = lane & 15;
  const int mb = (lane >> 4) << 3;
  const int mrow0 = m0 + wave * 16 + mb;
#pragma unroll
  for (int ns = 0; ns < 4; ++ns) {
    int gn = n0 + ns * 16 + cn;
    float bv = bp ? bp[gn] : 0.f;
#pragma unroll
    for (int r = 0; r < 8; ++r) {
      int gm = mrow0 + r;
      if (gm >= M) continue;
      float v = acc[ns][r] + bv;
      if (act_gelu) v = gelu_f(v);
      if (mode == 1) {
        int bb = gm / N_;
        v += meas_xy[(size_t)gm * 2]     * meas_W[gn]
           + meas_xy[(size_t)gm * 2 + 1] * meas_W[D_ + gn]
           + meas_v[gm]                  * meas_W[2 * D_ + gn]
           + meas_b[gn];
        C[((size_t)gm + bb) * Nn + gn] = v;   // row remap bn -> b*S + n
      } else {
        size_t oi = (size_t)gm * Nn + gn;
        if (resid) v += resid[oi];
        C[oi] = v;
      }
    }
  }
}

// ---------------------------------------------------------------------------
// Patch gather: pf[bn][0..80] = 9x9 zero-padded city patch, cols 81..95 = 0
// ---------------------------------------------------------------------------
__global__ void patch_gather_kernel(const float* __restrict__ meas_xy,
                                    const float* __restrict__ city,
                                    float* __restrict__ pf)
{
  int idx = blockIdx.x * blockDim.x + threadIdx.x;
  if (idx >= B_ * N_ * PP_) return;
  int c = idx % PP_;
  int bn = idx / PP_;
  float v = 0.f;
  if (c < P_ * P_) {
    int b = bn / N_;
    float mx = meas_xy[(size_t)bn * 2];
    float my = meas_xy[(size_t)bn * 2 + 1];
    int x = (int)roundf(mx * (W_ - 1)); x = min(max(x, 0), W_ - 1);
    int y = (int)roundf(my * (H_ - 1)); y = min(max(y, 0), H_ - 1);
    int row = c / P_, col = c % P_;
    int yy = y + row - R_, xx = x + col - R_;
    if (yy >= 0 && yy < H_ && xx >= 0 && xx < W_)
      v = city[((size_t)b * H_ + yy) * W_ + xx];
  }
  pf[idx] = v;
}

__global__ void bs_token_kernel(const float* __restrict__ bs_xy, const float* __restrict__ bs_W,
                                const float* __restrict__ bs_b, float* __restrict__ src)
{
  int b = blockIdx.x, d = threadIdx.x;
  float v = bs_xy[b * 2] * bs_W[d] + bs_xy[b * 2 + 1] * bs_W[D_ + d] + bs_b[d];
  src[((size_t)b * S_ + N_) * D_ + d] = v;
}

__global__ void task_vec_kernel(const float* __restrict__ task_emb, const float* __restrict__ taskp_W,
                                const float* __restrict__ taskp_b, const int* __restrict__ task_id,
                                float* __restrict__ tvec)
{
  int b = blockIdx.x, d = threadIdx.x;
  int t = task_id[b];
  float s = taskp_b[d];
  for (int j = 0; j < D_; ++j) s += task_emb[t * D_ + j] * taskp_W[j * D_ + d];
  tvec[b * D_ + d] = s;
}

__global__ void grid_init_kernel(const float* __restrict__ grid_pos, const float* __restrict__ tvec,
                                 float* __restrict__ grid)
{
  size_t idx = (size_t)blockIdx.x * blockDim.x + threadIdx.x;
  if (idx >= (size_t)B_ * K_ * D_) return;
  int d = (int)(idx & (D_ - 1));
  int b = (int)(idx >> 20);                  // K_*D_ = 2^20
  grid[idx] = grid_pos[idx & ((size_t)K_ * D_ - 1)] + tvec[b * D_ + d];
}

// one wave32 per row, D=256
__launch_bounds__(256)
__global__ void layernorm_kernel(const float* __restrict__ x, const float* __restrict__ g,
                                 const float* __restrict__ bta, float* __restrict__ y, int rows)
{
  int wave = threadIdx.x >> 5, lane = threadIdx.x & 31;
  int row = blockIdx.x * 8 + wave;
  if (row >= rows) return;
  const float* xr = x + (size_t)row * D_;
  float vals[8];
  float s = 0.f;
#pragma unroll
  for (int i = 0; i < 8; ++i) { vals[i] = xr[lane + i * 32]; s += vals[i]; }
#pragma unroll
  for (int off = 16; off; off >>= 1) s += __shfl_xor(s, off, 32);
  float mean = s * (1.f / 256.f);
  float vs = 0.f;
#pragma unroll
  for (int i = 0; i < 8; ++i) { float d = vals[i] - mean; vs += d * d; }
#pragma unroll
  for (int off = 16; off; off >>= 1) vs += __shfl_xor(vs, off, 32);
  float inv = rsqrtf(vs * (1.f / 256.f) + 1e-5f);
#pragma unroll
  for (int i = 0; i < 8; ++i) {
    int c = lane + i * 32;
    y[(size_t)row * D_ + c] = (vals[i] - mean) * inv * g[c] + bta[c];
  }
}

// density[b,k] = mean_n exp(-|gxy[k]-meas_xy[b,n]|^2 / (2 sigma^2)); one wave per (b,k)
__launch_bounds__(256)
__global__ void density_kernel(const float* __restrict__ meas_xy, float* __restrict__ dens)
{
  int wave = threadIdx.x >> 5, lane = threadIdx.x & 31;
  int idx = blockIdx.x * 8 + wave;
  if (idx >= B_ * K_) return;
  int b = idx >> 12, k = idx & (K_ - 1);
  float gx = ((k & 63) + 0.5f) * (1.f / 64.f);
  float gy = ((k >> 6) + 0.5f) * (1.f / 64.f);
  float s = 0.f;
  for (int n = lane; n < N_; n += 32) {
    float dx = gx - meas_xy[((size_t)b * N_ + n) * 2];
    float dy = gy - meas_xy[((size_t)b * N_ + n) * 2 + 1];
    s += expf(-(dx * dx + dy * dy) * INV2SIG2_);
  }
#pragma unroll
  for (int off = 16; off; off >>= 1) s += __shfl_xor(s, off, 32);
  if (lane == 0) dens[idx] = s * (1.f / N_);
}

// ---------------------------------------------------------------------------
// Fused attention: one block per (b, head, 16 grid rows). Scores via WMMA,
// bias/gate computed on the fly, stable softmax in LDS, attn@V via WMMA with
// K-dim split across 4 waves, reduced with LDS float atomics.
// ---------------------------------------------------------------------------
__launch_bounds__(128)
__global__ void attention_kernel(const float* __restrict__ Q, const float* __restrict__ Kc,
                                 const float* __restrict__ Vc,
                                 const float* __restrict__ meas_xy, const float* __restrict__ bs_xy,
                                 const float* __restrict__ dens,
                                 const float* __restrict__ dist_raw, const float* __restrict__ dens_raw,
                                 int layer, float* __restrict__ Out)
{
  __shared__ float scores[16][S_];     // 32 KB
  __shared__ float red[16][8];
  __shared__ float outacc[2][16][16];
  const int tid = threadIdx.x;
  const int wave = tid >> 5, lane = tid & 31;
  const int kt = blockIdx.x, h = blockIdx.y, b = blockIdx.z;
  const int krow0 = kt * 16;
  const int cn = lane & 15, mb = (lane >> 4) << 3;

  const float sp = log1pf(expf(dist_raw[layer]));   // softplus
  const float th = tanhf(dens_raw[layer]);

  // ---- scores = Q @ K^T : M=16, N=512, Kd=HD=32 (one WMMA k-step) ----
  FragH ua;
  {
    const float* qp = Q + ((size_t)b * K_ + krow0 + (lane & 15)) * D_ + h * HD_;
    int kb = (lane >> 4) << 3;
#pragma unroll
    for (int i = 0; i < 2; ++i) {
      float4 t0 = *(const float4*)(qp + kb + i * 4);
      float4 t1 = *(const float4*)(qp + 16 + kb + i * 4);
      ua.e[i * 4 + 0] = (_Float16)t0.x; ua.e[i * 4 + 1] = (_Float16)t0.y;
      ua.e[i * 4 + 2] = (_Float16)t0.z; ua.e[i * 4 + 3] = (_Float16)t0.w;
      ua.e[8 + i * 4 + 0] = (_Float16)t1.x; ua.e[8 + i * 4 + 1] = (_Float16)t1.y;
      ua.e[8 + i * 4 + 2] = (_Float16)t1.z; ua.e[8 + i * 4 + 3] = (_Float16)t1.w;
    }
  }
#pragma unroll
  for (int ns = 0; ns < 8; ++ns) {
    int n0 = wave * 128 + ns * 16;
    FragH ub;   // B[kdim=hd][n=s]: per-lane K row s, k 8h..8h+7 then 16+8h..16+8h+7 (h = lane/16)
    const float* kp = Kc + ((size_t)b * S_ + n0 + (lane & 15)) * D_ + h * HD_ + ((lane >> 4) << 3);
#pragma unroll
    for (int i = 0; i < 4; ++i) {
      float4 t = *(const float4*)(kp + i * 4 + (i >> 1) * 8);
      ub.e[i * 4 + 0] = (_Float16)t.x; ub.e[i * 4 + 1] = (_Float16)t.y;
      ub.e[i * 4 + 2] = (_Float16)t.z; ub.e[i * 4 + 3] = (_Float16)t.w;
    }
    v8f c0 = {};
    c0 = __builtin_amdgcn_wmma_f32_16x16x32_f16(false, ua.v, false, ub.v, (short)0, c0, false, false);
#pragma unroll
    for (int r = 0; r < 8; ++r) scores[mb + r][n0 + cn] = c0[r];
  }
  __syncthreads();

  // ---- scale + distance bias + density gate (on the fly) ----
  for (int i = tid; i < 16 * S_; i += 128) {
    int r = i >> 9, sc = i & (S_ - 1);
    int kg = krow0 + r;
    float gx = ((kg & 63) + 0.5f) * (1.f / 64.f);
    float gy = (((kg >> 6) & 63) + 0.5f) * (1.f / 64.f);
    float sx, sy;
    if (sc < N_) {
      sx = meas_xy[((size_t)b * N_ + sc) * 2];
      sy = meas_xy[((size_t)b * N_ + sc) * 2 + 1];
    } else {
      sx = bs_xy[b * 2]; sy = bs_xy[b * 2 + 1];
    }
    float dx = gx - sx, dy = gy - sy;
    float biasv = -sqrtf(dx * dx + dy * dy);
    float gate = 1.f + th * dens[(size_t)b * K_ + kg];
    scores[r][sc] = (scores[r][sc] * SCALE_ + sp * biasv) * gate;
  }
  __syncthreads();

  // ---- softmax over s (8 threads per row) ----
  {
    int r = tid >> 3, sub = tid & 7;
    float mx = -3.4e38f;
    for (int c = sub; c < S_; c += 8) mx = fmaxf(mx, scores[r][c]);
    red[r][sub] = mx;
    __syncthreads();
#pragma unroll
    for (int j = 0; j < 8; ++j) mx = fmaxf(mx, red[r][j]);
    float sm = 0.f;
    for (int c = sub; c < S_; c += 8) { float e = expf(scores[r][c] - mx); scores[r][c] = e; sm += e; }
    __syncthreads();
    red[r][sub] = sm;
    __syncthreads();
    sm = 0.f;
#pragma unroll
    for (int j = 0; j < 8; ++j) sm += red[r][j];
    float inv = 1.f / sm;
    for (int c = sub; c < S_; c += 8) scores[r][c] *= inv;
  }
  for (int i = tid; i < 2 * 16 * 16; i += 128) ((float*)outacc)[i] = 0.f;
  __syncthreads();

  // ---- out = attn @ V : M=16, N=32, Kd=512; each wave owns a 128-wide K slab ----
  v8f oacc[2] = {};
#pragma unroll
  for (int kk = 0; kk < 4; ++kk) {
    int k0 = wave * 128 + kk * 32;
    FragH pa;
    {
      int rowa = lane & 15;
      int kb = (lane >> 4) << 3;
#pragma unroll
      for (int i = 0; i < 2; ++i) {
        float4 t0 = *(const float4*)(&scores[rowa][k0 + kb + i * 4]);
        float4 t1 = *(const float4*)(&scores[rowa][k0 + 16 + kb + i * 4]);
        pa.e[i * 4 + 0] = (_Float16)t0.x; pa.e[i * 4 + 1] = (_Float16)t0.y;
        pa.e[i * 4 + 2] = (_Float16)t0.z; pa.e[i * 4 + 3] = (_Float16)t0.w;
        pa.e[8 + i * 4 + 0] = (_Float16)t1.x; pa.e[8 + i * 4 + 1] = (_Float16)t1.y;
        pa.e[8 + i * 4 + 2] = (_Float16)t1.z; pa.e[8 + i * 4 + 3] = (_Float16)t1.w;
      }
    }
#pragma unroll
    for (int ns = 0; ns < 2; ++ns) {
      FragH vbf;  // B[kdim=s][n=hd] (strided in global, scalar loads)
      const float* vp = Vc + ((size_t)b * S_ + k0 + ((lane >> 4) << 3)) * D_
                        + h * HD_ + ns * 16 + (lane & 15);
#pragma unroll
      for (int i = 0; i < 16; ++i) vbf.e[i] = (_Float16)vp[(size_t)(i + (i >> 3) * 8) * D_];
      oacc[ns] = __builtin_amdgcn_wmma_f32_16x16x32_f16(false, pa.v, false, vbf.v,
                                                        (short)0, oacc[ns], false, false);
    }
  }
#pragma unroll
  for (int ns = 0; ns < 2; ++ns)
#pragma unroll
    for (int r = 0; r < 8; ++r)
      atomicAdd(&outacc[ns][mb + r][cn], oacc[ns][r]);
  __syncthreads();

  for (int i = tid; i < 512; i += 128) {
    int ns = i >> 8, r = (i >> 4) & 15, c = i & 15;
    Out[((size_t)b * K_ + krow0 + r) * D_ + h * HD_ + ns * 16 + c] = outacc[ns][r][c];
  }
}

// logits[row] = dot(tf[row], head_W) + head_b
__launch_bounds__(256)
__global__ void head_kernel(const float* __restrict__ tf, const float* __restrict__ head_W,
                            const float* __restrict__ head_b, float* __restrict__ out)
{
  int wave = threadIdx.x >> 5, lane = threadIdx.x & 31;
  int row = blockIdx.x * 8 + wave;
  if (row >= B_ * K_) return;
  const float* tp = tf + (size_t)row * D_;
  float s = 0.f;
#pragma unroll
  for (int i = 0; i < 8; ++i) s += tp[lane + i * 32] * head_W[lane + i * 32];
#pragma unroll
  for (int off = 16; off; off >>= 1) s += __shfl_xor(s, off, 32);
  if (lane == 0) out[row] = s + head_b[0];
}

// ---------------------------------------------------------------------------
extern "C" void kernel_launch(void* const* d_in, const int* in_sizes, int n_in,
                              void* d_out, int out_size, void* d_ws, size_t ws_size,
                              hipStream_t stream)
{
  (void)in_sizes; (void)n_in; (void)out_size; (void)ws_size;
  const float* meas_xy  = (const float*)d_in[0];
  const float* meas_v   = (const float*)d_in[1];
  const float* bs_xy    = (const float*)d_in[2];
  const float* city     = (const float*)d_in[3];
  const float* meas_W   = (const float*)d_in[4];
  const float* meas_b   = (const float*)d_in[5];
  const float* patch_W1 = (const float*)d_in[6];
  const float* patch_b1 = (const float*)d_in[7];
  const float* patch_W2 = (const float*)d_in[8];
  const float* patch_b2 = (const float*)d_in[9];
  const float* bs_W     = (const float*)d_in[10];
  const float* bs_b     = (const float*)d_in[11];
  const float* task_emb = (const float*)d_in[12];
  const float* taskp_W  = (const float*)d_in[13];
  const float* taskp_b  = (const float*)d_in[14];
  const float* grid_pos = (const float*)d_in[15];
  const float* lnq_g    = (const float*)d_in[16];
  const float* lnq_b    = (const float*)d_in[17];
  const float* lnkv_g   = (const float*)d_in[18];
  const float* lnkv_b   = (const float*)d_in[19];
  const float* qW       = (const float*)d_in[20];
  const float* qb       = (const float*)d_in[21];
  const float* kW       = (const float*)d_in[22];
  const float* kb       = (const float*)d_in[23];
  const float* vW       = (const float*)d_in[24];
  const float* vb       = (const float*)d_in[25];
  const float* oW       = (const float*)d_in[26];
  const float* ob       = (const float*)d_in[27];
  const float* dist_raw = (const float*)d_in[28];
  const float* dens_raw = (const float*)d_in[29];
  const float* lnf_g    = (const float*)d_in[30];
  const float* lnf_b    = (const float*)d_in[31];
  const float* f1W      = (const float*)d_in[32];
  const float* f1b      = (const float*)d_in[33];
  const float* f2W      = (const float*)d_in[34];
  const float* f2b      = (const float*)d_in[35];
  const float* tfp_W    = (const float*)d_in[36];
  const float* tfp_b    = (const float*)d_in[37];
  const float* head_W   = (const float*)d_in[38];
  const float* head_b   = (const float*)d_in[39];
  const int*   task_id  = (const int*)d_in[40];

  float* ws = (float*)d_ws;
  size_t o = 0;
  float* srcb  = ws + o; o += (size_t)B_ * S_ * D_;       // tokens (B,S,D)
  float* pf    = ws + o; o += (size_t)B_ * N_ * PP_;      // padded patches
  float* H1    = ws + o; o += (size_t)B_ * N_ * D_;       // patch hidden
  float* gridb = ws + o; o += (size_t)B_ * K_ * D_;       // grid state
  float* scr   = ws + o; o += (size_t)B_ * K_ * D_;       // LN out / attn out
  float* Qb    = ws + o; o += (size_t)B_ * K_ * D_;       // Q / task-proj tf
  float* kvl   = ws + o; o += (size_t)B_ * S_ * D_;       // LN(kv)
  float* Kb    = ws + o; o += (size_t)B_ * S_ * D_;
  float* Vb    = ws + o; o += (size_t)B_ * S_ * D_;
  float* hid   = ws + o; o += (size_t)B_ * K_ * 2 * D_;   // FFN hidden
  float* dens  = ws + o; o += (size_t)B_ * K_;
  float* tvec  = ws + o; o += (size_t)B_ * D_;

  const int BN = B_ * N_, BK = B_ * K_, BS = B_ * S_;
  dim3 blk(128);

  patch_gather_kernel<<<(BN * PP_ + 255) / 256, 256, 0, stream>>>(meas_xy, city, pf);
  // H1 = gelu(pf @ patch_W1 + b1)   (K padded 81 -> 96)
  gemm_kernel<<<dim3(D_ / 64, (BN + 63) / 64), blk, 0, stream>>>(
      pf, patch_W1, patch_b1, H1, nullptr, BN, D_, PP_, 81, 1, 0,
      nullptr, nullptr, nullptr, nullptr, nullptr);
  // src(meas rows) = H1 @ patch_W2 + b2 + meas-linear (fused, remapped to b*S+n)
  gemm_kernel<<<dim3(D_ / 64, (BN + 63) / 64), blk, 0, stream>>>(
      H1, patch_W2, patch_b2, srcb, nullptr, BN, D_, D_, D_, 0, 1,
      meas_xy, meas_v, meas_W, meas_b, nullptr);
  bs_token_kernel<<<B_, D_, 0, stream>>>(bs_xy, bs_W, bs_b, srcb);
  task_vec_kernel<<<B_, D_, 0, stream>>>(task_emb, taskp_W, taskp_b, task_id, tvec);
  grid_init_kernel<<<(BK * D_ + 255) / 256, 256, 0, stream>>>(grid_pos, tvec, gridb);
  density_kernel<<<(BK + 7) / 8, 256, 0, stream>>>(meas_xy, dens);

  for (int l = 0; l < 2; ++l) {
    layernorm_kernel<<<(BK + 7) / 8, 256, 0, stream>>>(gridb, lnq_g + l * D_, lnq_b + l * D_, scr, BK);
    gemm_kernel<<<dim3(D_ / 64, BK / 64), blk, 0, stream>>>(
        scr, qW + (size_t)l * D_ * D_, qb + l * D_, Qb, nullptr, BK, D_, D_, D_, 0, 0,
        nullptr, nullptr, nullptr, nullptr, nullptr);
    layernorm_kernel<<<(BS + 7) / 8, 256, 0, stream>>>(srcb, lnkv_g + l * D_, lnkv_b + l * D_, kvl, BS);
    gemm_kernel<<<dim3(D_ / 64, BS / 64), blk, 0, stream>>>(
        kvl, kW + (size_t)l * D_ * D_, kb + l * D_, Kb, nullptr, BS, D_, D_, D_, 0, 0,
        nullptr, nullptr, nullptr, nullptr, nullptr);
    gemm_kernel<<<dim3(D_ / 64, BS / 64), blk, 0, stream>>>(
        kvl, vW + (size_t)l * D_ * D_, vb + l * D_, Vb, nullptr, BS, D_, D_, D_, 0, 0,
        nullptr, nullptr, nullptr, nullptr, nullptr);
    attention_kernel<<<dim3(K_ / 16, HEADS_, B_), blk, 0, stream>>>(
        Qb, Kb, Vb, meas_xy, bs_xy, dens, dist_raw, dens_raw, l, scr);
    // grid += attn_out @ oW + ob
    gemm_kernel<<<dim3(D_ / 64, BK / 64), blk, 0, stream>>>(
        scr, oW + (size_t)l * D_ * D_, ob + l * D_, gridb, gridb, BK, D_, D_, D_, 0, 0,
        nullptr, nullptr, nullptr, nullptr, nullptr);
    layernorm_kernel<<<(BK + 7) / 8, 256, 0, stream>>>(gridb, lnf_g + l * D_, lnf_b + l * D_, scr, BK);
    gemm_kernel<<<dim3(2 * D_ / 64, BK / 64), blk, 0, stream>>>(
        scr, f1W + (size_t)l * D_ * 2 * D_, f1b + l * 2 * D_, hid, nullptr,
        BK, 2 * D_, D_, D_, 1, 0, nullptr, nullptr, nullptr, nullptr, nullptr);
    gemm_kernel<<<dim3(D_ / 64, BK / 64), blk, 0, stream>>>(
        hid, f2W + (size_t)l * 2 * D_ * D_, f2b + l * D_, gridb, gridb,
        BK, D_, 2 * D_, 2 * D_, 0, 0, nullptr, nullptr, nullptr, nullptr, nullptr);
  }
  // tf = gelu(grid @ tfp_W[task_id[b]] + tfp_b[task_id[b]])  (reuse Qb)
  gemm_kernel<<<dim3(D_ / 64, BK / 64), blk, 0, stream>>>(
      gridb, tfp_W, tfp_b, Qb, nullptr, BK, D_, D_, D_, 1, 2,
      nullptr, nullptr, nullptr, nullptr, task_id);
  head_kernel<<<(BK + 7) / 8, 256, 0, stream>>>(Qb, head_W, head_b, (float*)d_out);
}
